// VVAttention_84284438216883
// MI455X (gfx1250) — matrix operand order repaired and checked by reading, not compiled, not run on hardware
//
#include <hip/hip_runtime.h>
#include <hip/hip_bf16.h>
#include <hip/hip_fp16.h>

// VV (cosine-similarity) attention for MI455X / gfx1250, wave32 + WMMA +
// async-LDS (ASYNCcnt) double-buffered pipelines.
//
// Roofline: ~35.5 GFLOP vs ~53MB HBM traffic -> compute bound; all three
// matmuls run on v_wmma_f32_16x16x32_f16 (fp32 accum). fp32 inputs are
// converted to f16 once in an elementwise pass so the GEMM K-loop moves only
// f16 tiles (async DMA) and does no per-element cvt work.

typedef __attribute__((ext_vector_type(16))) _Float16 v16h;
typedef __attribute__((ext_vector_type(4)))  _Float16 v4h;
typedef __attribute__((ext_vector_type(8)))  float    v8f;

#define N_SEQ  1024
#define BATCH  8
#define DMODEL 768
#define NHEADS 12
#define HDIM   64
#define ATTN_SCALE 0.125f   // hd^-0.5 * SCALE_MULT

// ---- CDNA5 async copy helpers (inline asm; portable across toolchains) ----
// vdst VGPR carries the LDS byte address (ISA 10.x; an offset from the LDS base);
// the low 32 bits of a generic pointer to __shared__ are that LDS offset.
__device__ __forceinline__ void async_load_b128(uint32_t lds_addr, const void* gaddr) {
  asm volatile("global_load_async_to_lds_b128 %0, %1, off"
               :: "v"(lds_addr), "v"((uint64_t)(uintptr_t)gaddr)
               : "memory");
}
template <int N>
__device__ __forceinline__ void s_wait_async() {
  asm volatile("s_wait_asynccnt %0" :: "n"(N) : "memory");
}
__device__ __forceinline__ uint32_t lds_addr_of(const void* p) {
  return (uint32_t)(uintptr_t)p;
}

// A-fragment k index for 16-bit 16x32 A (contiguous, matches B fills):
//   lanes 0-15 : elems 0-15 -> K=j
//   lanes 16-31: elems 0-15 -> K=j+16
__device__ __forceinline__ int a_kmap(int j, int hi) {
  return j + (hi << 4);
}

// ---------------------------------------------------------------------------
// Kernel 0: one-shot fp32 -> f16 conversion (pure bandwidth, ~2.5us total).
// ---------------------------------------------------------------------------
__global__ __launch_bounds__(256)
void cvt_f32_to_f16(const float* __restrict__ src, _Float16* __restrict__ dst) {
  const size_t i = ((size_t)blockIdx.x * blockDim.x + threadIdx.x) * 4;
  float4 v = *reinterpret_cast<const float4*>(src + i);
  v4h o = { (_Float16)v.x, (_Float16)v.y, (_Float16)v.z, (_Float16)v.w };
  *reinterpret_cast<v4h*>(dst + i) = o;
}

// ---------------------------------------------------------------------------
// Kernel 1: V = X @ Wv^T  (X[b,n,:] = query[n,b,:]) + per-head-row L2 norms.
// 64x64 output tile per block == one full head. A/B f16 tiles are async-DMA'd
// into double-buffered LDS; fragment gathers are pure ds_load_b128.
// ---------------------------------------------------------------------------
__global__ __launch_bounds__(128)
void vv_gemm_v(const _Float16* __restrict__ qh, const _Float16* __restrict__ wh,
               _Float16* __restrict__ Vh, _Float16* __restrict__ Vnh) {
  __shared__ __align__(16) _Float16 As[2][64 * 64];   // act tile    (f16, 8KB)
  __shared__ __align__(16) _Float16 Bs[2][64 * 64];   // weight tile (f16, 8KB)
  __shared__ __align__(16) float    tile[64 * 64];    // fp32 output staging
  __shared__ float rnorm[64];

  const int tid  = threadIdx.x;
  const int wave = tid >> 5;
  const int lane = tid & 31;
  const int lo   = lane & 15;
  const int hi   = lane >> 4;

  const int h        = blockIdx.y;           // head = 64-col tile
  const int row_base = blockIdx.x * 64;      // row in [B*N]
  const int b        = row_base >> 10;
  const int n_base   = row_base & 1023;
  const int col_base = h * HDIM;

  // 8KB f16 tile = 512 b128 chunks = 4 per thread per tile (A + B = 8).
  auto issue_stage = [&](int kt, int buf) {
    const int kkb = kt * 64 * 2;             // byte offset of k-slice start
    const uint32_t lA = lds_addr_of(&As[buf][0]);
    const uint32_t lB = lds_addr_of(&Bs[buf][0]);
#pragma unroll
    for (int i = 0; i < 4; ++i) {
      const uint32_t off = (uint32_t)(i * 128 + tid) * 16u;  // byte in tile
      const int r  = (int)(off >> 7);        // row (128B per 64-half row)
      const int cb = (int)(off & 127);       // byte within row
      const char* ga = (const char*)(qh + ((size_t)(n_base + r) * BATCH + b) * DMODEL) + kkb + cb;
      const char* gb = (const char*)(wh + (size_t)(col_base + r) * DMODEL) + kkb + cb;
      async_load_b128(lA + off, ga);
      async_load_b128(lB + off, gb);
    }
  };

  v8f c[4] = {};
  issue_stage(0, 0);

  for (int kt = 0; kt < DMODEL / 64; ++kt) {
    const int buf = kt & 1;
    if (kt < DMODEL / 64 - 1) {
      issue_stage(kt + 1, buf ^ 1);
      s_wait_async<8>();    // 8 outstanding = only next stage in flight
    } else {
      s_wait_async<0>();
    }
    __syncthreads();

    const _Float16* A  = &As[buf][0];
    const _Float16* Bt = &Bs[buf][0];
#pragma unroll
    for (int kh = 0; kh < 2; ++kh) {
      const int kk2 = kh * 32;
      v16h a;
#pragma unroll
      for (int j = 0; j < 16; ++j)
        a[j] = A[(wave * 16 + lo) * 64 + kk2 + a_kmap(j, hi)];
#pragma unroll
      for (int t = 0; t < 4; ++t) {
        v16h bf;
#pragma unroll
        for (int j = 0; j < 16; ++j)
          bf[j] = Bt[(t * 16 + lo) * 64 + kk2 + hi * 16 + j];
        c[t] = __builtin_amdgcn_wmma_f32_16x16x32_f16(false, a, false, bf,
                                                      (short)0, c[t], false, false);
      }
    }
    __syncthreads();
  }

  // Epilogue: spill C into LDS, row norms, dual f16 writeout.
#pragma unroll
  for (int t = 0; t < 4; ++t)
#pragma unroll
    for (int r = 0; r < 8; ++r)
      tile[(wave * 16 + r + hi * 8) * 64 + t * 16 + lo] = c[t][r];
  __syncthreads();

  if (tid < 64) {
    float s = 0.f;
#pragma unroll 8
    for (int d = 0; d < HDIM; ++d) {
      float v = tile[tid * 64 + d];
      s += v * v;
    }
    rnorm[tid] = rsqrtf(s);
  }
  __syncthreads();

  const size_t obase = ((size_t)(b * NHEADS + h) * N_SEQ + n_base) * HDIM;
#pragma unroll 4
  for (int i = 0; i < 32; ++i) {
    int idx = i * 128 + tid;
    int r = idx >> 6;
    float v = tile[idx];
    Vh[obase + idx]  = (_Float16)v;
    Vnh[obase + idx] = (_Float16)(v * rnorm[r]);
  }
}

// ---------------------------------------------------------------------------
// Kernel 2: flash attention per (b,h). 64 q-rows per block; K/V f16 tiles
// async-DMA'd into double-buffered LDS; online softmax; P round-trips through
// wave-private LDS (C-layout -> A-layout) for the P@V WMMAs.
// ---------------------------------------------------------------------------
__global__ __launch_bounds__(128)
void vv_attn(const _Float16* __restrict__ Vh, const _Float16* __restrict__ Vnh,
             float* __restrict__ out) {
  __shared__ __align__(16) _Float16 Qs[64 * 64];
  __shared__ __align__(16) _Float16 Ks[2][64 * 64];
  __shared__ __align__(16) _Float16 Vs[2][64 * 64];
  __shared__ __align__(16) _Float16 Ps[4][16 * 64];

  const int tid  = threadIdx.x;
  const int wave = tid >> 5;
  const int lane = tid & 31;
  const int lo   = lane & 15;
  const int hi   = lane >> 4;

  const int q0 = blockIdx.x * 64;
  const int bh = blockIdx.y;
  const int b  = bh / NHEADS;
  const int h  = bh % NHEADS;

  const size_t base = (size_t)bh * N_SEQ * HDIM;   // [b,h] slab, hd-major

  // 8KB f16 tile = 512 b128 chunks = 4 per thread per tile.
  auto issue_kv = [&](int j0, int buf) {
    const uint32_t lK = lds_addr_of(&Ks[buf][0]);
    const uint32_t lV = lds_addr_of(&Vs[buf][0]);
    const char* gK = (const char*)(Vnh + base + (size_t)j0 * HDIM);
    const char* gV = (const char*)(Vh  + base + (size_t)j0 * HDIM);
#pragma unroll
    for (int i = 0; i < 4; ++i) {
      const uint32_t off = (uint32_t)(i * 128 + tid) * 16u;
      async_load_b128(lK + off, gK + off);
      async_load_b128(lV + off, gV + off);
    }
  };

  // Preamble: async-copy Q tile + first K/V stage, then wait/barrier once.
  {
    const uint32_t lQ = lds_addr_of(Qs);
    const char* gQ = (const char*)(Vnh + base + (size_t)q0 * HDIM);
#pragma unroll
    for (int i = 0; i < 4; ++i) {
      const uint32_t off = (uint32_t)(i * 128 + tid) * 16u;
      async_load_b128(lQ + off, gQ + off);
    }
  }
  issue_kv(0, 0);
  s_wait_async<0>();
  __syncthreads();

  // Q A-fragments (rows = lane&15 of this wave's 16-row strip), K-dim 0..63.
  v16h aq0, aq1;
#pragma unroll
  for (int j = 0; j < 16; ++j) {
    int k = a_kmap(j, hi);
    aq0[j] = Qs[(wave * 16 + lo) * 64 + k];
    aq1[j] = Qs[(wave * 16 + lo) * 64 + 32 + k];
  }

  v8f o[4] = {};
  float mi[8], li[8];
#pragma unroll
  for (int r = 0; r < 8; ++r) { mi[r] = -1e30f; li[r] = 0.f; }

  for (int kt = 0; kt < N_SEQ / 64; ++kt) {
    const int buf = kt & 1;
    if (kt < N_SEQ / 64 - 1) {
      issue_kv((kt + 1) * 64, buf ^ 1);
      s_wait_async<8>();    // 8 outstanding = only next K/V stage in flight
    } else {
      s_wait_async<0>();
    }
    __syncthreads();

    const _Float16* K = &Ks[buf][0];
    const _Float16* V = &Vs[buf][0];

    // S = Qn * Kn^T  (B[k=d, n=key] gathered from row-major K[key][d]).
    v8f s[4];
#pragma unroll
    for (int t = 0; t < 4; ++t) {
      v16h bk0, bk1;
#pragma unroll
      for (int j = 0; j < 16; ++j) {
        int kr = (t * 16 + lo) * 64 + hi * 16 + j;
        bk0[j] = K[kr];
        bk1[j] = K[kr + 32];
      }
      v8f z = {};
      s[t] = __builtin_amdgcn_wmma_f32_16x16x32_f16(false, aq0, false, bk0,
                                                    (short)0, z, false, false);
      s[t] = __builtin_amdgcn_wmma_f32_16x16x32_f16(false, aq1, false, bk1,
                                                    (short)0, s[t], false, false);
    }

    // Online softmax. C-layout: VGPR r = row r (lanes 0-15) / r+8 (16-31);
    // row values live in one 16-lane group -> shfl_xor masks {1,2,4,8}.
#pragma unroll
    for (int r = 0; r < 8; ++r) {
      float rmax = fmaxf(fmaxf(s[0][r], s[1][r]), fmaxf(s[2][r], s[3][r])) * ATTN_SCALE;
#pragma unroll
      for (int m = 1; m < 16; m <<= 1) rmax = fmaxf(rmax, __shfl_xor(rmax, m, 32));
      float newm = fmaxf(mi[r], rmax);
      float corr = __expf(mi[r] - newm);
      float rsum = 0.f;
#pragma unroll
      for (int t = 0; t < 4; ++t) {
        float p = __expf(s[t][r] * ATTN_SCALE - newm);
        s[t][r] = p;
        rsum += p;
      }
#pragma unroll
      for (int m = 1; m < 16; m <<= 1) rsum += __shfl_xor(rsum, m, 32);
      li[r] = li[r] * corr + rsum;
      mi[r] = newm;
#pragma unroll
      for (int t = 0; t < 4; ++t) o[t][r] *= corr;
    }

    // P: C-layout -> f16 in wave-private LDS (row-major [16][64]).
#pragma unroll
    for (int t = 0; t < 4; ++t)
#pragma unroll
      for (int r = 0; r < 8; ++r)
        Ps[wave][(r + hi * 8) * 64 + t * 16 + lo] = (_Float16)s[t][r];

    // P A-fragments (K-dim = key index).
    v16h ap0, ap1;
#pragma unroll
    for (int j = 0; j < 16; ++j) {
      int k = a_kmap(j, hi);
      ap0[j] = Ps[wave][lo * 64 + k];
      ap1[j] = Ps[wave][lo * 64 + 32 + k];
    }

    // O += P @ V  (B[k=key, n=d] from row-major V[key][d]).
#pragma unroll
    for (int t = 0; t < 4; ++t) {
      v16h bv0, bv1;
#pragma unroll
      for (int j = 0; j < 16; ++j) {
        int key = hi * 16 + j;
        bv0[j] = V[key * 64 + t * 16 + lo];
        bv1[j] = V[(key + 32) * 64 + t * 16 + lo];
      }
      o[t] = __builtin_amdgcn_wmma_f32_16x16x32_f16(false, ap0, false, bv0,
                                                    (short)0, o[t], false, false);
      o[t] = __builtin_amdgcn_wmma_f32_16x16x32_f16(false, ap1, false, bv1,
                                                    (short)0, o[t], false, false);
    }
    __syncthreads();
  }

  // Epilogue: normalize by li, write out[n, b, h*64 + d] (fp32).
#pragma unroll
  for (int r = 0; r < 8; ++r) {
    float inv = 1.0f / li[r];
    int n = q0 + wave * 16 + r + hi * 8;
#pragma unroll
    for (int t = 0; t < 4; ++t) {
      int d = t * 16 + lo;
      out[((size_t)n * BATCH + b) * DMODEL + h * HDIM + d] = o[t][r] * inv;
    }
  }
}

// ---------------------------------------------------------------------------
extern "C" void kernel_launch(void* const* d_in, const int* in_sizes, int n_in,
                              void* d_out, int out_size, void* d_ws, size_t ws_size,
                              hipStream_t stream) {
  (void)in_sizes; (void)n_in; (void)out_size; (void)ws_size;
  const float* query = (const float*)d_in[0];   // [N, B, D] fp32
  const float* wv    = (const float*)d_in[1];   // [D, D]    fp32
  float* out = (float*)d_out;                   // [N, B, D] fp32

  // Workspace layout (f16): Vh, Vnh [B,H,N,hd] (12.58MB each),
  //                         Qf16 [N,B,D] (12.58MB), Wf16 [D,D] (1.18MB).
  const size_t nV = (size_t)BATCH * NHEADS * N_SEQ * HDIM;   // 6.29M elems
  const size_t nQ = (size_t)N_SEQ * BATCH * DMODEL;          // 6.29M elems
  const size_t nW = (size_t)DMODEL * DMODEL;                 // 0.59M elems
  _Float16* Vh   = (_Float16*)d_ws;
  _Float16* Vnh  = Vh + nV;
  _Float16* Qf16 = Vnh + nV;
  _Float16* Wf16 = Qf16 + nQ;

  cvt_f32_to_f16<<<(int)(nQ / 4 / 256), 256, 0, stream>>>(query, Qf16);
  cvt_f32_to_f16<<<(int)(nW / 4 / 256), 256, 0, stream>>>(wv, Wf16);
  vv_gemm_v<<<dim3(128, NHEADS), 128, 0, stream>>>(Qf16, Wf16, Vh, Vnh);
  vv_attn<<<dim3(N_SEQ / 64, BATCH * NHEADS), 128, 0, stream>>>(Vh, Vnh, out);
}
